// CausalSelfAttention_30880814858965
// MI455X (gfx1250) — matrix operand order repaired and checked by reading, not compiled, not run on hardware
//
#include <hip/hip_runtime.h>
#include <hip/hip_bf16.h>
#include <hip/hip_fp16.h>

typedef __attribute__((ext_vector_type(16))) _Float16 v16h;
typedef __attribute__((ext_vector_type(8)))  _Float16 v8h;
typedef __attribute__((ext_vector_type(4)))  _Float16 v4h;
typedef __attribute__((ext_vector_type(8)))  float    v8f;
typedef unsigned int u32x4 __attribute__((ext_vector_type(4)));
typedef int          i32x4 __attribute__((ext_vector_type(4)));
typedef int          i32x8 __attribute__((ext_vector_type(8)));

__device__ inline v8f zero8() {
    v8f z = {0.0f,0.0f,0.0f,0.0f,0.0f,0.0f,0.0f,0.0f};
    return z;
}

// DPP lane-permute move (row = 16 lanes). Pure VALU cross-lane, no LDS.
template <int CTRL>
__device__ inline float dpp_movf(float x) {
    return __int_as_float(__builtin_amdgcn_update_dpp(
        0, __float_as_int(x), CTRL, 0xF, 0xF, true));
}
__device__ inline float row16_max(float v) {
    v = fmaxf(v, dpp_movf<0xB1>(v));    // quad_perm(1,0,3,2)
    v = fmaxf(v, dpp_movf<0x4E>(v));    // quad_perm(2,3,0,1)
    v = fmaxf(v, dpp_movf<0x141>(v));   // row_half_mirror
    v = fmaxf(v, dpp_movf<0x140>(v));   // row_mirror
    return v;
}
__device__ inline float row16_sum(float v) {
    v += dpp_movf<0xB1>(v);
    v += dpp_movf<0x4E>(v);
    v += dpp_movf<0x141>(v);
    v += dpp_movf<0x140>(v);
    return v;
}

// ---------------------------------------------------------------------------
// Tensor Data Mover: DMA a 2D f16 tile (tileW contiguous elems x tileH rows,
// row stride rowStride elems) from global memory into LDS. TENSORcnt-tracked.
// ---------------------------------------------------------------------------
__device__ inline void tdm_load_2d_f16(unsigned lds_off, const _Float16* gptr,
                                       unsigned tileW, unsigned tileH,
                                       unsigned tensorW, unsigned tensorH,
                                       unsigned rowStride)
{
    unsigned long long ga = (unsigned long long)(size_t)gptr;
    u32x4 g0;
    g0[0] = 1u;                                    // count=1, user descriptor
    g0[1] = lds_off;                               // lds_addr
    g0[2] = (unsigned)(ga & 0xffffffffull);        // global_addr lo
    g0[3] = (unsigned)((ga >> 32) & 0x1ffffffull)  // global_addr hi
          | (2u << 30);                            // type=2 ("image")
    i32x8 g1;
    g1[0] = (int)(1u << 16);                                   // data_size=1 (2B)
    g1[1] = (int)((tensorW & 0xffffu) << 16);
    g1[2] = (int)((tensorW >> 16) & 0xffffu)
          | (int)((tensorH & 0xffffu) << 16);
    g1[3] = (int)((tensorH >> 16) & 0xffffu)
          | (int)((tileW & 0xffffu) << 16);                    // tile_dim0
    g1[4] = (int)(tileH & 0xffffu);                            // tile_dim1
    g1[5] = (int)rowStride;                                    // dim0_stride lo32
    g1[6] = 0;
    g1[7] = 0;
    i32x4 z4 = {0,0,0,0};
#if __clang_major__ >= 23
    i32x8 z8 = {0,0,0,0,0,0,0,0};
    __builtin_amdgcn_tensor_load_to_lds(g0, g1, z4, z4, z8, 0);
#else
    __builtin_amdgcn_tensor_load_to_lds(g0, g1, z4, z4, 0);
#endif
}

// ---------------------------------------------------------------------------
__global__ __launch_bounds__(256)
void convert_f16(const float* __restrict__ in, _Float16* __restrict__ out, int n4)
{
    int i = (blockIdx.x * 256 + threadIdx.x);
    if (i < n4) {
        const float4 f = ((const float4*)in)[i];
        v4h o = { (_Float16)f.x, (_Float16)f.y, (_Float16)f.z, (_Float16)f.w };
        ((v4h*)out)[i] = o;
    }
}

__global__ __launch_bounds__(256)
void transpose_f16(const float* __restrict__ W, _Float16* __restrict__ Wt,
                   int Kd, int N)
{
    __shared__ _Float16 t[32][33];
    const int n0 = blockIdx.x * 32, k0 = blockIdx.y * 32;
    const int tx = threadIdx.x & 31, tg = threadIdx.x >> 5;
    #pragma unroll
    for (int i = 0; i < 4; ++i) {
        int k = tg * 4 + i;
        t[tx][k] = (_Float16)W[(size_t)(k0 + k) * N + n0 + tx];
    }
    __syncthreads();
    #pragma unroll
    for (int i = 0; i < 4; ++i) {
        int nr = tg * 4 + i;
        Wt[(size_t)(n0 + nr) * Kd + k0 + tx] = t[nr][tx];
    }
}

// ---------------------------------------------------------------------------
// GEMM: out[M,N] = A[M,Kd] (f16) * Bt[N,Kd]^T (f16) + bias, f32 accumulate.
// Macro tile 256x128, 8 waves (4x2), wave tile 64x64 (4x4 WMMA).
// TDM-staged, double-buffered.
// ---------------------------------------------------------------------------
template <int OUT_F16>
__global__ __launch_bounds__(256)
void gemm_wmma_tdm(const _Float16* __restrict__ A, const _Float16* __restrict__ Bt,
                   const float* __restrict__ bias, void* __restrict__ out,
                   int M, int N, int Kd)
{
    __shared__ __align__(16) _Float16 lA[2][256][32];
    __shared__ __align__(16) _Float16 lB[2][128][32];

    const int tid   = threadIdx.x;
    const int wid   = tid >> 5;
    const int lane  = tid & 31;
    const int lhalf = lane >> 4;
    const int l16   = lane & 15;
    const int bm = blockIdx.x * 256;
    const int bn = blockIdx.y * 128;
    const int wm = (wid & 3) * 64;
    const int wn = (wid >> 2) * 64;

    v8f acc[4][4];
    #pragma unroll
    for (int i = 0; i < 4; ++i)
        #pragma unroll
        for (int j = 0; j < 4; ++j) acc[i][j] = zero8();

    const int nsteps = Kd >> 5;
    int cur = 0;

    if (wid == 0) {
        tdm_load_2d_f16((unsigned)(size_t)&lA[0][0][0], A + (size_t)bm * Kd,
                        32, 256, Kd, M, Kd);
        tdm_load_2d_f16((unsigned)(size_t)&lB[0][0][0], Bt + (size_t)bn * Kd,
                        32, 128, Kd, N, Kd);
        __builtin_amdgcn_s_wait_tensorcnt(0);
    }
    __syncthreads();

    for (int s = 0; s < nsteps; ++s) {
        const int nxt = cur ^ 1;
        if (wid == 0 && (s + 1) < nsteps) {
            const int k0 = (s + 1) << 5;
            tdm_load_2d_f16((unsigned)(size_t)&lA[nxt][0][0],
                            A + (size_t)bm * Kd + k0, 32, 256, Kd, M, Kd);
            tdm_load_2d_f16((unsigned)(size_t)&lB[nxt][0][0],
                            Bt + (size_t)bn * Kd + k0, 32, 128, Kd, N, Kd);
        }

        v16h af[4], bf[4];
        #pragma unroll
        for (int mt = 0; mt < 4; ++mt) {
            const _Float16* rowp = &lA[cur][wm + mt*16 + l16][0];
            v8h c0 = *(const v8h*)(rowp + 8*lhalf);
            v8h c1 = *(const v8h*)(rowp + 16 + 8*lhalf);
            #pragma unroll
            for (int e = 0; e < 8; ++e) { af[mt][e] = c0[e]; af[mt][e+8] = c1[e]; }
        }
        #pragma unroll
        for (int nt = 0; nt < 4; ++nt) {
            const _Float16* rowp = &lB[cur][wn + nt*16 + l16][0];
            v8h c0 = *(const v8h*)(rowp + 8*lhalf);
            v8h c1 = *(const v8h*)(rowp + 16 + 8*lhalf);
            #pragma unroll
            for (int e = 0; e < 8; ++e) { bf[nt][e] = c0[e]; bf[nt][e+8] = c1[e]; }
        }
        #pragma unroll
        for (int mt = 0; mt < 4; ++mt)
            #pragma unroll
            for (int nt = 0; nt < 4; ++nt)
                acc[mt][nt] = __builtin_amdgcn_wmma_f32_16x16x32_f16(
                    false, af[mt], false, bf[nt], (short)0, acc[mt][nt], false, false);

        if (wid == 0) __builtin_amdgcn_s_wait_tensorcnt(0);
        __syncthreads();
        cur = nxt;
    }

    #pragma unroll
    for (int nt = 0; nt < 4; ++nt) {
        const int n  = bn + wn + nt*16 + l16;
        const float bv = bias[n];
        #pragma unroll
        for (int mt = 0; mt < 4; ++mt) {
            const size_t mb = (size_t)(bm + wm + mt*16) * N + n;
            #pragma unroll
            for (int r = 0; r < 8; ++r) {
                const float v = acc[mt][nt][r] + bv;
                const size_t idx = mb + (size_t)(r + 8*lhalf) * N;
                if (OUT_F16) ((_Float16*)out)[idx] = (_Float16)v;
                else         ((float*)out)[idx]    = v;
            }
        }
    }
}

// ---------------------------------------------------------------------------
// Flash-style causal attention. qkv: f16 [B*K, 3C]. y: f16 [B*K, C].
// Grid: (Kseq/128, B*H), 128 threads = 4 waves; each wave owns TWO 16-row
// Q tiles (32 rows), so staged lK/lVt tiles and kf/vf fragments are reused
// across two A-operands: 16 WMMA per wave per 32-wide KV chunk.
// ---------------------------------------------------------------------------
__global__ __launch_bounds__(128)
void attn_wmma(const _Float16* __restrict__ qkv, _Float16* __restrict__ y,
               int Kseq, int C, int H)
{
    const int Dh   = C / H;          // 64
    const int bh   = blockIdx.y;
    const int b    = bh / H, h = bh % H;
    const int tid  = threadIdx.x;
    const int wid  = tid >> 5, lane = tid & 31;
    const int lhalf = lane >> 4, l16 = lane & 15;
    const int qb0   = blockIdx.x * 128;
    const int qbw   = qb0 + wid * 32;        // wave's 32 query rows
    const int row3C = 3 * C;

    const _Float16* Qp = qkv + (size_t)b * Kseq * row3C + (size_t)h * Dh;
    const _Float16* Kp = Qp + C;
    const _Float16* Vp = Qp + 2 * C;

    __shared__ __align__(16) _Float16 lK[32][64];       // K rows [j][d]  (TDM)
    __shared__ __align__(16) _Float16 lVt[64][32];      // V transposed [d][j]
    __shared__ __align__(16) _Float16 lP[8][16][32];    // per (wave,qt) P tile

    // Q fragments (A layout): two q-tiles x two 32-wide k-slices
    v16h qf[2][2];
    #pragma unroll
    for (int qt = 0; qt < 2; ++qt) {
        const _Float16* qrow = Qp + (size_t)(qbw + qt*16 + l16) * row3C;
        #pragma unroll
        for (int kk = 0; kk < 2; ++kk) {
            v8h c0 = *(const v8h*)(qrow + kk*32 + 8*lhalf);
            v8h c1 = *(const v8h*)(qrow + kk*32 + 16 + 8*lhalf);
            #pragma unroll
            for (int e = 0; e < 8; ++e) { qf[qt][kk][e] = c0[e]; qf[qt][kk][e+8] = c1[e]; }
        }
    }

    v8f o[2][4];
    float mrow[2][8], lrow[2][8];
    #pragma unroll
    for (int qt = 0; qt < 2; ++qt) {
        #pragma unroll
        for (int dt = 0; dt < 4; ++dt) o[qt][dt] = zero8();
        #pragma unroll
        for (int r = 0; r < 8; ++r) { mrow[qt][r] = -3.0e38f; lrow[qt][r] = 0.0f; }
    }

    const int jend = qb0 + 128;
    for (int jb = 0; jb < jend; jb += 32) {
        // K tile via Tensor Data Mover (32 rows x 64 halves, stride 3C)
        if (wid == 0)
            tdm_load_2d_f16((unsigned)(size_t)&lK[0][0],
                            Kp + (size_t)jb * row3C,
                            64, 32, (unsigned)row3C, (unsigned)Kseq,
                            (unsigned)row3C);
        // V tile, transposed by hand (TDM cannot transpose)
        {
            int j  = tid >> 2;
            int d0 = (tid & 3) * 16;
            const _Float16* vrow = Vp + (size_t)(jb + j) * row3C + d0;
            v8h a0 = *(const v8h*)(vrow);
            v8h a1 = *(const v8h*)(vrow + 8);
            #pragma unroll
            for (int e = 0; e < 8; ++e) { lVt[d0+e][j] = a0[e]; lVt[d0+8+e][j] = a1[e]; }
        }
        if (wid == 0) __builtin_amdgcn_s_wait_tensorcnt(0);
        __syncthreads();

        // Batch-load ALL K fragments, then run the 8 S-WMMAs
        v16h kf[2][2];
        #pragma unroll
        for (int nt = 0; nt < 2; ++nt) {
            const _Float16* krow = &lK[nt*16 + l16][0];
            #pragma unroll
            for (int kk = 0; kk < 2; ++kk) {
                v8h c0 = *(const v8h*)(krow + kk*32 + 8*lhalf);
                v8h c1 = *(const v8h*)(krow + kk*32 + 16 + 8*lhalf);
                #pragma unroll
                for (int e = 0; e < 8; ++e) { kf[nt][kk][e] = c0[e]; kf[nt][kk][e+8] = c1[e]; }
            }
        }
        v8f s[2][2];
        #pragma unroll
        for (int qt = 0; qt < 2; ++qt)
            #pragma unroll
            for (int nt = 0; nt < 2; ++nt) {
                v8f c = __builtin_amdgcn_wmma_f32_16x16x32_f16(
                            false, qf[qt][0], false, kf[nt][0], (short)0, zero8(), false, false);
                s[qt][nt] = __builtin_amdgcn_wmma_f32_16x16x32_f16(
                            false, qf[qt][1], false, kf[nt][1], (short)0, c, false, false);
            }

        // scale + causal mask + online softmax (DPP row reductions)
        #pragma unroll
        for (int qt = 0; qt < 2; ++qt) {
            #pragma unroll
            for (int nt = 0; nt < 2; ++nt) {
                int j = jb + nt*16 + l16;
                #pragma unroll
                for (int r = 0; r < 8; ++r) {
                    int q = qbw + qt*16 + r + 8*lhalf;
                    float v = s[qt][nt][r] * 0.125f;
                    s[qt][nt][r] = (j <= q) ? v : -3.0e38f;
                }
            }
            float mnew[8], alpha[8];
            #pragma unroll
            for (int r = 0; r < 8; ++r) {
                float v = row16_max(fmaxf(s[qt][0][r], s[qt][1][r]));
                mnew[r]  = fmaxf(mrow[qt][r], v);
                alpha[r] = __expf(mrow[qt][r] - mnew[r]);
            }
            #pragma unroll
            for (int nt = 0; nt < 2; ++nt)
                #pragma unroll
                for (int r = 0; r < 8; ++r)
                    s[qt][nt][r] = __expf(s[qt][nt][r] - mnew[r]);
            #pragma unroll
            for (int r = 0; r < 8; ++r) {
                float v = row16_sum(s[qt][0][r] + s[qt][1][r]);
                lrow[qt][r] = lrow[qt][r] * alpha[r] + v;
                mrow[qt][r] = mnew[r];
            }
            #pragma unroll
            for (int dt = 0; dt < 4; ++dt)
                #pragma unroll
                for (int r = 0; r < 8; ++r) o[qt][dt][r] *= alpha[r];

            // P (C layout) -> LDS (A-fragment staging)
            #pragma unroll
            for (int nt = 0; nt < 2; ++nt)
                #pragma unroll
                for (int r = 0; r < 8; ++r)
                    lP[wid*2 + qt][r + 8*lhalf][nt*16 + l16] = (_Float16)s[qt][nt][r];
        }
        __syncthreads();

        // Batch-load V fragments + P fragments, then the 8 PV-WMMAs
        v16h vf[4];
        #pragma unroll
        for (int dt = 0; dt < 4; ++dt) {
            const _Float16* vrow = &lVt[dt*16 + l16][0];
            v8h c0 = *(const v8h*)(vrow + 8*lhalf);
            v8h c1 = *(const v8h*)(vrow + 16 + 8*lhalf);
            #pragma unroll
            for (int e = 0; e < 8; ++e) { vf[dt][e] = c0[e]; vf[dt][e+8] = c1[e]; }
        }
        v16h pf[2];
        #pragma unroll
        for (int qt = 0; qt < 2; ++qt) {
            const _Float16* prow = &lP[wid*2 + qt][l16][0];
            v8h c0 = *(const v8h*)(prow + 8*lhalf);
            v8h c1 = *(const v8h*)(prow + 16 + 8*lhalf);
            #pragma unroll
            for (int e = 0; e < 8; ++e) { pf[qt][e] = c0[e]; pf[qt][e+8] = c1[e]; }
        }
        #pragma unroll
        for (int qt = 0; qt < 2; ++qt)
            #pragma unroll
            for (int dt = 0; dt < 4; ++dt)
                o[qt][dt] = __builtin_amdgcn_wmma_f32_16x16x32_f16(
                                false, pf[qt], false, vf[dt], (short)0, o[qt][dt], false, false);
        __syncthreads();
    }

    // Normalize and write y (f16)
    #pragma unroll
    for (int qt = 0; qt < 2; ++qt) {
        float inv[8];
        #pragma unroll
        for (int r = 0; r < 8; ++r) inv[r] = 1.0f / lrow[qt][r];
        #pragma unroll
        for (int dt = 0; dt < 4; ++dt) {
            int d = h * Dh + dt*16 + l16;
            #pragma unroll
            for (int r = 0; r < 8; ++r) {
                int q = qbw + qt*16 + r + 8*lhalf;
                y[((size_t)b * Kseq + q) * C + d] = (_Float16)(o[qt][dt][r] * inv[r]);
            }
        }
    }
}

// ---------------------------------------------------------------------------
extern "C" void kernel_launch(void* const* d_in, const int* in_sizes, int n_in,
                              void* d_out, int out_size, void* d_ws, size_t ws_size,
                              hipStream_t stream) {
    const float* x     = (const float*)d_in[0];
    const float* Wqkv  = (const float*)d_in[1];
    const float* bqkv  = (const float*)d_in[2];
    const float* Wproj = (const float*)d_in[3];
    const float* bproj = (const float*)d_in[4];

    const int Kseq = 2048;
    const int C    = 1024;
    const int BK   = in_sizes[0] / C;      // B*K = 4096
    const int H    = 16;
    const int C3   = 3 * C;

    char* w = (char*)d_ws;
    _Float16* xh     = (_Float16*)w;                      w += (size_t)BK * C  * 2;
    _Float16* Wqkvt  = (_Float16*)w;                      w += (size_t)C3 * C  * 2;
    _Float16* Wprojt = (_Float16*)w;                      w += (size_t)C  * C  * 2;
    _Float16* qkvh   = (_Float16*)w;                      w += (size_t)BK * C3 * 2;
    _Float16* yh     = (_Float16*)w;

    {
        int n4 = (BK * C) / 4;
        convert_f16<<<(n4 + 255) / 256, 256, 0, stream>>>(x, xh, n4);
        dim3 gq(C3 / 32, C / 32);
        transpose_f16<<<gq, 256, 0, stream>>>(Wqkv, Wqkvt, C, C3);
        dim3 gp(C / 32, C / 32);
        transpose_f16<<<gp, 256, 0, stream>>>(Wproj, Wprojt, C, C);
    }
    {
        dim3 grid(BK / 256, C3 / 128);
        gemm_wmma_tdm<1><<<grid, 256, 0, stream>>>(xh, Wqkvt, bqkv, qkvh, BK, C3, C);
    }
    {
        dim3 grid(Kseq / 128, (BK / Kseq) * H);
        attn_wmma<<<grid, 128, 0, stream>>>(qkvh, yh, Kseq, C, H);
    }
    {
        dim3 grid(BK / 256, C / 128);
        gemm_wmma_tdm<0><<<grid, 256, 0, stream>>>(yh, Wprojt, bproj, d_out, BK, C, C);
    }
}
